// Attention_Bahdanau_17085379903495
// MI455X (gfx1250) — matrix operand order repaired and checked by reading, not compiled, not run on hardware
//
#include <hip/hip_runtime.h>

typedef __attribute__((ext_vector_type(16))) __bf16 v16bf;
typedef __attribute__((ext_vector_type(8)))  float  v8f;

#define AS3 __attribute__((address_space(3)))

// fast tanh: hardware v_tanh_f32 on gfx1250, clamped exp fallback otherwise
__device__ __forceinline__ float fast_tanh(float x) {
#if __has_builtin(__builtin_amdgcn_tanh_f32)
    return __builtin_amdgcn_tanh_f32(x);
#else
    x = fminf(fmaxf(x, -15.0f), 15.0f);
    float e = __expf(2.0f * x);
    return (e - 1.0f) / (e + 1.0f);
#endif
}

__device__ __forceinline__ void async_ld_b128(unsigned lds_addr, const void* gaddr) {
    // GLOBAL_LOAD_ASYNC_TO_LDS_B128: 16 B global -> LDS copy, tracked by ASYNCcnt
    asm volatile("global_load_async_to_lds_b128 %0, %1, off"
                 :: "v"(lds_addr), "v"(gaddr) : "memory");
}
__device__ __forceinline__ void async_ld_b128_off16(unsigned lds_addr, const void* gaddr) {
    asm volatile("global_load_async_to_lds_b128 %0, %1, off offset:16"
                 :: "v"(lds_addr), "v"(gaddr) : "memory");
}
__device__ __forceinline__ void wait_asynccnt0() {
#if __has_builtin(__builtin_amdgcn_s_wait_asynccnt)
    __builtin_amdgcn_s_wait_asynccnt(0);
#else
    asm volatile("s_wait_asynccnt 0x0" ::: "memory");
#endif
}

// ---------------------------------------------------------------------------
// Kernel 0: pack W_bot (rows D..2D-1 of W, layout W[k][n], x@W) into bf16
// WMMA-B tiles. Tile (nt, kt) is 32(K) x 16(N). Lane l holds 16 contiguous
// bf16 (32 B): n = nt*16 + (l&15), k = kt*32 + (l>>4)*16 + j, j = 0..15.
// ---------------------------------------------------------------------------
__global__ void pack_wbot_kernel(const float* __restrict__ W,
                                 __bf16* __restrict__ wsB, int D) {
    int gid = blockIdx.x * blockDim.x + threadIdx.x;
    if (gid >= D * D) return;
    const int KT = D >> 5;
    int j    =  gid        & 15;
    int lane = (gid >> 4)  & 31;
    int t9   =  gid >> 9;                  // nt*KT + kt
    int kt   =  t9 % KT;
    int nt   =  t9 / KT;
    int k = (kt << 5) + ((lane >> 4) << 4) + j;
    int n = (nt << 4) + (lane & 15);
    wsB[gid] = (__bf16)W[(size_t)(D + k) * D + n];
}

// ---------------------------------------------------------------------------
// Kernel 1: hb[b][n] = bias[n] + sum_k hidden[b][k] * W_top[k][n]   (f32)
// ---------------------------------------------------------------------------
__global__ void hb_kernel(const float* __restrict__ hidden,
                          const float* __restrict__ W,
                          const float* __restrict__ bias,
                          float* __restrict__ hb, int D) {
    int n = blockIdx.x * blockDim.x + threadIdx.x;
    int b = blockIdx.y;
    const float* h = hidden + (size_t)b * D;
    float acc = bias[n];
    for (int k = 0; k < D; ++k) acc = fmaf(h[k], W[(size_t)k * D + n], acc);
    hb[(size_t)b * D + n] = acc;
}

// ---------------------------------------------------------------------------
// Kernel 2: fused  scores[b][s] = v . tanh( enc[b,s,:] @ Wbot + hb[b,:] )
// 8 waves / block; wave w owns a 16-row M tile; N in groups of 8 tiles with
// full-K f32 accumulation (v_wmma_f32_16x16x32_bf16). B tiles staged to a
// double-buffered LDS via async direct-to-LDS loads: one barrier per K-step,
// DMA overlapped with WMMA.
// ---------------------------------------------------------------------------
__launch_bounds__(256)
__global__ void scores_kernel(const float* __restrict__ enc,    // [B,S,D] f32
                              const float* __restrict__ hb,     // [B,D] bias folded
                              const float* __restrict__ vvec,   // [D]
                              const __bf16* __restrict__ wsB,   // packed B tiles
                              float* __restrict__ scores,       // [B,S]
                              int S, int D) {
    const int b    = blockIdx.y;
    const int wave = threadIdx.x >> 5;
    const int lane = threadIdx.x & 31;
    const int m_base = blockIdx.x * 128 + wave * 16;
    const int g  = lane >> 4;          // half-wave group
    const int ml = lane & 15;          // A row / B,C column within tile

    const int KT  = D >> 5;            // 32-wide K tiles
    const int NTB = D >> 7;            // groups of 8 N tiles

    const float* rowp = enc + ((size_t)b * S + (size_t)(m_base + ml)) * D;

    __shared__ __attribute__((aligned(16))) __bf16 ldsB[2][8 * 512];  // 16 KB

    // LDS byte address of this wave's staging slot (as3 ptr -> 32-bit LDS addr)
    const unsigned lds_base =
        (unsigned)(size_t)(AS3 char*)(void*)&ldsB[0][0];
    const unsigned lds_my = lds_base + (unsigned)(wave * 1024 + lane * 32);

    float p[8];
#pragma unroll
    for (int r = 0; r < 8; ++r) p[r] = 0.0f;

    for (int ntb = 0; ntb < NTB; ++ntb) {
        // this wave stages N-tile nt = ntb*8 + wave for every kt
        const char* gsrc = (const char*)wsB +
            ((size_t)((ntb * 8 + wave) * KT) << 10) + (unsigned)(lane * 32);

        v8f acc[8];
#pragma unroll
        for (int tt = 0; tt < 8; ++tt) acc[tt] = v8f{};

        // prologue: stage kt=0 into buffer 0
        async_ld_b128(lds_my, gsrc);
        async_ld_b128_off16(lds_my, gsrc);
        wait_asynccnt0();
        __syncthreads();

        for (int kt = 0; kt < KT; ++kt) {
            const int cur = kt & 1;
            // stage kt+1 into the other buffer (its last readers finished
            // before the previous barrier)
            if (kt + 1 < KT) {
                const char* gn = gsrc + ((size_t)(kt + 1) << 10);
                unsigned dn = lds_my + (unsigned)((1 - cur) << 13);
                async_ld_b128(dn, gn);
                async_ld_b128_off16(dn, gn);
            }

            // Build A fragment (16x32 bf16): lane half g reads K run
            // {kc+16g..+15} of its row, f32 -> bf16.
            const int kc = kt << 5;
            const float4* r0 = (const float4*)(rowp + kc + (g << 4));
            const float4* r1 = (const float4*)(rowp + kc + 8 + (g << 4));
            float4 f0 = r0[0], f1 = r0[1];
            float4 f2 = r1[0], f3 = r1[1];
            if (kt + 1 < KT) __builtin_prefetch(rowp + kc + 32, 0, 1);

            v16bf av;
            av[0]=(__bf16)f0.x;  av[1]=(__bf16)f0.y;  av[2]=(__bf16)f0.z;  av[3]=(__bf16)f0.w;
            av[4]=(__bf16)f1.x;  av[5]=(__bf16)f1.y;  av[6]=(__bf16)f1.z;  av[7]=(__bf16)f1.w;
            av[8]=(__bf16)f2.x;  av[9]=(__bf16)f2.y;  av[10]=(__bf16)f2.z; av[11]=(__bf16)f2.w;
            av[12]=(__bf16)f3.x; av[13]=(__bf16)f3.y; av[14]=(__bf16)f3.z; av[15]=(__bf16)f3.w;

            const __bf16* bb = &ldsB[cur][0];
#pragma unroll
            for (int tt = 0; tt < 8; ++tt) {
                v16bf bv = *(const v16bf*)(bb + (tt * 512 + lane * 16));
                acc[tt] = __builtin_amdgcn_wmma_f32_16x16x32_bf16(
                    false, av, false, bv, (short)0, acc[tt], false, false);
            }

            wait_asynccnt0();     // own stage of kt+1 landed in LDS
            __syncthreads();      // everyone's stage visible, reads of cur done
        }

        // Epilogue: tanh + v-dot for the 8 completed N tiles.
        // C layout: lane -> N (=ml), VGPR r -> M = r + 8*g.
#pragma unroll
        for (int tt = 0; tt < 8; ++tt) {
            int n = (((ntb << 3) + tt) << 4) + ml;
            float hn = hb[(size_t)b * D + n];
            float vn = vvec[n];
#pragma unroll
            for (int r = 0; r < 8; ++r)
                p[r] = fmaf(fast_tanh(acc[tt][r] + hn), vn, p[r]);
        }
    }

    // Reduce over the 16 lanes of each half-wave (distinct n per lane).
#pragma unroll
    for (int r = 0; r < 8; ++r) {
        float x = p[r];
        x += __shfl_xor(x, 1, 32);
        x += __shfl_xor(x, 2, 32);
        x += __shfl_xor(x, 4, 32);
        x += __shfl_xor(x, 8, 32);
        p[r] = x;
    }
    if (ml == 0) {
#pragma unroll
        for (int r = 0; r < 8; ++r)
            scores[(size_t)b * S + m_base + (g << 3) + r] = p[r];
    }
}

// ---------------------------------------------------------------------------
// Kernel 3: masked softmax over S per batch. mask: s >= src_len[b] -> -1e18
// ---------------------------------------------------------------------------
__global__ void softmax_kernel(const float* __restrict__ scores,
                               const int* __restrict__ src_len,
                               float* __restrict__ out, int S) {
    const int b = blockIdx.x;
    const int t = threadIdx.x;          // 256 threads, S/256 == 8 per thread
    const int len = src_len[b];

    float vals[8];
    float m = -3.0e38f;
#pragma unroll
    for (int i = 0; i < 8; ++i) {
        int s = t + (i << 8);
        float x = scores[(size_t)b * S + s];
        if (s >= len) x = -1.0e18f;
        vals[i] = x;
        m = fmaxf(m, x);
    }

    __shared__ float red[8];
    for (int mk = 16; mk >= 1; mk >>= 1) m = fmaxf(m, __shfl_xor(m, mk, 32));
    if ((t & 31) == 0) red[t >> 5] = m;
    __syncthreads();
    float bm = red[0];
#pragma unroll
    for (int i = 1; i < 8; ++i) bm = fmaxf(bm, red[i]);
    __syncthreads();

    float sum = 0.0f;
#pragma unroll
    for (int i = 0; i < 8; ++i) {
        vals[i] = expf(vals[i] - bm);
        sum += vals[i];
    }
    for (int mk = 16; mk >= 1; mk >>= 1) sum += __shfl_xor(sum, mk, 32);
    if ((t & 31) == 0) red[t >> 5] = sum;
    __syncthreads();
    float bs = 0.0f;
#pragma unroll
    for (int i = 0; i < 8; ++i) bs += red[i];
    float inv = 1.0f / bs;

#pragma unroll
    for (int i = 0; i < 8; ++i) {
        int s = t + (i << 8);
        out[(size_t)b * S + s] = vals[i] * inv;
    }
}

// ---------------------------------------------------------------------------
extern "C" void kernel_launch(void* const* d_in, const int* in_sizes, int n_in,
                              void* d_out, int out_size, void* d_ws, size_t ws_size,
                              hipStream_t stream) {
    const float* hidden  = (const float*)d_in[0];   // [B,D]
    const float* enc     = (const float*)d_in[1];   // [B,S,D]
    const float* W       = (const float*)d_in[2];   // [2D,D]
    const float* bias    = (const float*)d_in[3];   // [D]
    const float* vvec    = (const float*)d_in[4];   // [D]
    const int*   src_len = (const int*)d_in[5];     // [B]

    const int B = in_sizes[5];
    const int D = in_sizes[0] / B;
    const int S = in_sizes[1] / in_sizes[0];        // (B*S*D)/(B*D)

    // workspace: hb [B*D] f32 | packed Wbot [D*D] bf16 | scores [B*S] f32
    float*  ws_hb = (float*)d_ws;
    __bf16* ws_B  = (__bf16*)(ws_hb + (size_t)B * D);
    float*  ws_sc = (float*)(ws_B + (size_t)D * D);

    pack_wbot_kernel<<<(D * D + 255) / 256, 256, 0, stream>>>(W, ws_B, D);
    hb_kernel<<<dim3(D / 256, B), 256, 0, stream>>>(hidden, W, bias, ws_hb, D);
    scores_kernel<<<dim3(S / 128, B), 256, 0, stream>>>(enc, ws_hb, vvec, ws_B,
                                                        ws_sc, S, D);
    softmax_kernel<<<B, 256, 0, stream>>>(ws_sc, src_len, (float*)d_out, S);
}
